// TemporalGCN4_28724741276164
// MI455X (gfx1250) — matrix operand order repaired and checked by reading, not compiled, not run on hardware
//
#include <hip/hip_runtime.h>

// ---------- types ----------
typedef __bf16 bf16_t;
typedef __attribute__((ext_vector_type(16))) __bf16 v16bf;
typedef __attribute__((ext_vector_type(8)))  __bf16 v8bf;
typedef __attribute__((ext_vector_type(8)))  float   v8f;

// ---------- device helpers ----------
__device__ __forceinline__ float wave_sum(float v) {
#pragma unroll
  for (int off = 16; off > 0; off >>= 1) v += __shfl_xor(v, off, 32);
  return v;
}
// order-preserving float<->uint encoding for atomicMax on floats
__device__ __forceinline__ unsigned f2ord(float f) {
  unsigned u = __float_as_uint(f);
  return (u & 0x80000000u) ? ~u : (u | 0x80000000u);
}
__device__ __forceinline__ float ord2f(unsigned u) {
  return __uint_as_float((u & 0x80000000u) ? (u & 0x7fffffffu) : ~u);
}

// ---------- elementwise ----------
__global__ void to_bf16_kernel(const float* __restrict__ src, bf16_t* __restrict__ dst, long long n) {
  long long i = (long long)blockIdx.x * blockDim.x + threadIdx.x;
  if (i < n) dst[i] = (bf16_t)src[i];
}
__global__ void fill_kernel(float* __restrict__ p, float v, long long n) {
  long long i = (long long)blockIdx.x * blockDim.x + threadIdx.x;
  if (i < n) p[i] = v;
}
__global__ void leaky_kernel(const float* __restrict__ in, float* __restrict__ out, long long n) {
  long long i = (long long)blockIdx.x * blockDim.x + threadIdx.x;
  if (i < n) { float x = in[i]; out[i] = x > 0.0f ? x : 0.01f * x; }
}
__global__ void init_nodes_kernel(unsigned* __restrict__ enc, float* __restrict__ denom, int N) {
  int i = blockIdx.x * blockDim.x + threadIdx.x;
  if (i < N) { enc[i] = f2ord(-INFINITY); denom[i] = 0.0f; }
}
__global__ void fix_amax_kernel(const unsigned* __restrict__ enc, float* __restrict__ amax, int N) {
  int i = blockIdx.x * blockDim.x + threadIdx.x;
  if (i < N) { float a = ord2f(enc[i]); amax[i] = __builtin_isfinite(a) ? a : 0.0f; }
}

// ---------- WMMA GEMM:  C[M,Nout] = A[M,K](bf16) * W[Nout,K]^T(bf16) + bias ----------
// One wave computes a 16x64 tile (4 accumulators along N): the A fragment is
// loaded once per K-step and reused by 4 v_wmma ops (4x less A traffic).
// M multiple of 16, Nout multiple of 64, K multiple of 32.
// A-fragment (16x32 bf16): lane m=lane&15, half=lane>>4; elems 0..7  -> K = half*8  + j
//                                                        elems 8..15 -> K = 16+half*8 + j
// B-fragment (32x16 bf16 = W^T): lane n=lane&15, half;   elems 0..7 -> K = half*8 + j, elems 8..15 -> K = 16+half*8 + j
// C/D layout: VGPR r, lane L: M = (L>>4)*8 + r, N = L&15.
__global__ void __launch_bounds__(256)
wmma_gemm_bias(const bf16_t* __restrict__ A, const bf16_t* __restrict__ W,
               const float* __restrict__ bias, float* __restrict__ C,
               int M, int K, int Nout) {
  const int lane  = threadIdx.x & 31;
  const int wave  = threadIdx.x >> 5;
  const int tilesM = M >> 4;
  const int tilesN = Nout >> 6;                  // groups of 64 columns
  const int tile = blockIdx.x * 8 + wave;
  if (tile >= tilesM * tilesN) return;           // whole-wave uniform: EXEC stays all-1s
  const int mTile = tile % tilesM;
  const int nG    = tile / tilesM;
  const int half  = lane >> 4;
  const int l16   = lane & 15;
  const bf16_t* aBase = A + (size_t)(mTile * 16 + l16) * K;
  const bf16_t* bBase0 = W + (size_t)(nG * 64 + l16) * K;        // sub-tile 0 column
  v8f acc0 = {}, acc1 = {}, acc2 = {}, acc3 = {};
  for (int kk = 0; kk < K; kk += 32) {
    v8bf a0 = *(const v8bf*)(aBase + kk + half * 8);
    v8bf a1 = *(const v8bf*)(aBase + kk + 16 + half * 8);
    v16bf av;
#pragma unroll
    for (int j = 0; j < 8; ++j) { av[j] = a0[j]; av[j + 8] = a1[j]; }
#pragma unroll
    for (int sub = 0; sub < 4; ++sub) {
      const bf16_t* bB = bBase0 + (size_t)(sub * 16) * K + kk + half * 8;
      v8bf b0 = *(const v8bf*)(bB);
      v8bf b1 = *(const v8bf*)(bB + 16);
      v16bf bv;
#pragma unroll
      for (int j = 0; j < 8; ++j) { bv[j] = b0[j]; bv[j + 8] = b1[j]; }
      if (sub == 0)      acc0 = __builtin_amdgcn_wmma_f32_16x16x32_bf16(false, av, false, bv, (short)0, acc0, false, false);
      else if (sub == 1) acc1 = __builtin_amdgcn_wmma_f32_16x16x32_bf16(false, av, false, bv, (short)0, acc1, false, false);
      else if (sub == 2) acc2 = __builtin_amdgcn_wmma_f32_16x16x32_bf16(false, av, false, bv, (short)0, acc2, false, false);
      else               acc3 = __builtin_amdgcn_wmma_f32_16x16x32_bf16(false, av, false, bv, (short)0, acc3, false, false);
    }
  }
  const int mBase = mTile * 16 + half * 8;
#pragma unroll
  for (int sub = 0; sub < 4; ++sub) {
    const int nCol = nG * 64 + sub * 16 + l16;
    const float bb = bias ? bias[nCol] : 0.0f;
    const v8f* accp = (sub == 0) ? &acc0 : (sub == 1) ? &acc1 : (sub == 2) ? &acc2 : &acc3;
#pragma unroll
    for (int r = 0; r < 8; ++r)
      C[(size_t)(mBase + r) * Nout + nCol] = (*accp)[r] + bb;
  }
}

// ---------- GRU gates ----------
__global__ void gru_gate_kernel(const float* __restrict__ gi, const float* __restrict__ gh,
                                const float* __restrict__ hprev, float* __restrict__ hnew,
                                float* __restrict__ gruout, int N, int H) {
  int idx = blockIdx.x * blockDim.x + threadIdx.x;
  if (idx >= N * H) return;
  int n = idx / H, c = idx - n * H;
  size_t b = (size_t)n * 3 * H;
  float ir = gi[b + c], iz = gi[b + H + c], in_ = gi[b + 2 * H + c];
  float hr = gh[b + c], hz = gh[b + H + c], hn  = gh[b + 2 * H + c];
  float r = 1.0f / (1.0f + __expf(-(ir + hr)));
  float z = 1.0f / (1.0f + __expf(-(iz + hz)));
  float nn = tanhf(in_ + r * hn);
  float h = (1.0f - z) * nn + z * hprev[idx];
  hnew[idx] = h; gruout[idx] = h;
}

// ---------- edge kernels (wave per edge, H channels lane-parallel) ----------
__global__ void __launch_bounds__(256)
edge_alpha_kernel(const float* __restrict__ q, const float* __restrict__ k,
                  const float* __restrict__ We, const float* __restrict__ ea,
                  const int* __restrict__ src, const int* __restrict__ dst,
                  float* __restrict__ alpha, unsigned* __restrict__ amax_enc,
                  int E, int H, float scale) {
  int e = blockIdx.x * 8 + (threadIdx.x >> 5);
  if (e >= E) return;
  int lane = threadIdx.x & 31;
  int s = src[e], d = dst[e];
  float w = ea[e];
  float p = 0.0f;
#pragma unroll
  for (int c = lane; c < H; c += 32)
    p += q[(size_t)d * H + c] * (k[(size_t)s * H + c] + w * We[c]);
  p = wave_sum(p);
  if (lane == 0) {
    float a = p * scale;
    alpha[e] = a;
    atomicMax(&amax_enc[d], f2ord(a));
  }
}
__global__ void edge_exp_kernel(float* __restrict__ alpha_ex, const float* __restrict__ amax,
                                float* __restrict__ denom, const int* __restrict__ dst, int E) {
  int e = blockIdx.x * blockDim.x + threadIdx.x;
  if (e >= E) return;
  int d = dst[e];
  float ex = __expf(alpha_ex[e] - amax[d]);
  alpha_ex[e] = ex;
  atomicAdd(&denom[d], ex);
}
__global__ void __launch_bounds__(256)
edge_msg_kernel(const float* __restrict__ ex, const float* __restrict__ denom,
                const float* __restrict__ v, const float* __restrict__ We,
                const float* __restrict__ ea, const int* __restrict__ src,
                const int* __restrict__ dst, float* __restrict__ agg, int E, int H) {
  int e = blockIdx.x * 8 + (threadIdx.x >> 5);
  if (e >= E) return;
  int lane = threadIdx.x & 31;
  int s = src[e], d = dst[e];
  float coeff = ex[e] / (denom[d] + 1e-16f);
  float w = ea[e];
#pragma unroll
  for (int c = lane; c < H; c += 32)
    atomicAdd(&agg[(size_t)d * H + c], coeff * (v[(size_t)s * H + c] + w * We[c]));
}

// ---------- fused temporal attention (wave per node) ----------
__global__ void __launch_bounds__(256)
attn_kernel(const float* __restrict__ hstack, const float* __restrict__ aW,
            const float* __restrict__ ab, float* __restrict__ hattn,
            int N, int H, int T) {
  int node = blockIdx.x * 8 + (threadIdx.x >> 5);
  if (node >= N) return;
  int lane = threadIdx.x & 31;
  if (T > 8) T = 8;
  float s[8];
  for (int t = 0; t < T; ++t) {
    const float* hp = hstack + ((size_t)t * N + node) * H;
    float p = 0.0f;
#pragma unroll
    for (int c = lane; c < H; c += 32) p += hp[c] * aW[c];
    s[t] = wave_sum(p) + ab[0];
  }
  float mx = s[0];
  for (int t = 1; t < T; ++t) mx = fmaxf(mx, s[t]);
  float aw[8]; float sum = 0.0f;
  for (int t = 0; t < T; ++t) { aw[t] = __expf(s[t] - mx); sum += aw[t]; }
  for (int t = 0; t < T; ++t) aw[t] /= sum;
  bool used[8];
  for (int t = 0; t < T; ++t) used[t] = false;
  int ksel = T < 3 ? T : 3;
  for (int it = 0; it < ksel; ++it) {
    int bi = 0; float bv = -1.0f;
    for (int t = 0; t < T; ++t) if (!used[t] && aw[t] > bv) { bv = aw[t]; bi = t; }
    used[bi] = true;
  }
  float ssum = 0.0f;
  for (int t = 0; t < T; ++t) if (used[t]) ssum += aw[t];
  float wgt[8];
  for (int t = 0; t < T; ++t) wgt[t] = used[t] ? aw[t] / (ssum + 1e-8f) : 0.0f;
#pragma unroll
  for (int c = lane; c < H; c += 32) {
    float acc = 0.0f;
    for (int t = 0; t < T; ++t) acc += wgt[t] * hstack[((size_t)t * N + node) * H + c];
    hattn[(size_t)node * H + c] = acc;
  }
}

// ---------- output conv (C_out = 1) ----------
__global__ void __launch_bounds__(256)
out_node_kernel(const float* __restrict__ hattn,
                const float* __restrict__ Wq, const float* __restrict__ Wk,
                const float* __restrict__ Wv, const float* __restrict__ Ws,
                const float* __restrict__ bq, const float* __restrict__ bk,
                const float* __restrict__ bv, const float* __restrict__ bs,
                float* __restrict__ q, float* __restrict__ k, float* __restrict__ v,
                float* __restrict__ out, int N, int H) {
  int node = blockIdx.x * 8 + (threadIdx.x >> 5);
  if (node >= N) return;
  int lane = threadIdx.x & 31;
  const float* hp = hattn + (size_t)node * H;
  float pq = 0, pk = 0, pv = 0, ps = 0;
#pragma unroll
  for (int c = lane; c < H; c += 32) {
    float x = hp[c];
    pq += x * Wq[c]; pk += x * Wk[c]; pv += x * Wv[c]; ps += x * Ws[c];
  }
  pq = wave_sum(pq); pk = wave_sum(pk); pv = wave_sum(pv); ps = wave_sum(ps);
  if (lane == 0) {
    q[node] = pq + bq[0]; k[node] = pk + bk[0]; v[node] = pv + bv[0];
    out[node] = ps + bs[0];   // skip term initializes output; messages atomically added
  }
}
__global__ void out_edge_alpha_kernel(const float* __restrict__ q, const float* __restrict__ k,
                                      const float* __restrict__ ea, const float* __restrict__ We,
                                      const int* __restrict__ src, const int* __restrict__ dst,
                                      float* __restrict__ alpha, unsigned* __restrict__ enc, int E) {
  int e = blockIdx.x * blockDim.x + threadIdx.x;
  if (e >= E) return;
  float a = q[dst[e]] * (k[src[e]] + ea[e] * We[0]);   // C=1 -> scale 1/sqrt(1)=1
  alpha[e] = a;
  atomicMax(&enc[dst[e]], f2ord(a));
}
__global__ void out_edge_msg_kernel(const float* __restrict__ ex, const float* __restrict__ denom,
                                    const float* __restrict__ v, const float* __restrict__ ea,
                                    const float* __restrict__ We, const int* __restrict__ src,
                                    const int* __restrict__ dst, float* __restrict__ out, int E) {
  int e = blockIdx.x * blockDim.x + threadIdx.x;
  if (e >= E) return;
  int d = dst[e];
  float coeff = ex[e] / (denom[d] + 1e-16f);
  atomicAdd(&out[d], coeff * (v[src[e]] + ea[e] * We[0]));
}

// ---------- host orchestration ----------
extern "C" void kernel_launch(void* const* d_in, const int* in_sizes, int n_in,
                              void* d_out, int out_size, void* d_ws, size_t ws_size,
                              hipStream_t stream) {
  const int T = 8, FIN = 32, H = 128, NL = 2;
  const int N = in_sizes[0] / (T * FIN);
  const int E = in_sizes[2] / T;
  const int G = 3 * H;
  const float scale = 0.08838834764831845f; // 1/sqrt(128)

  const float* x_seq   = (const float*)d_in[0];
  const int*   eis     = (const int*)  d_in[1];
  const float* eas     = (const float*)d_in[2];
  const float* gW_ih   = (const float*)d_in[3];
  const float* gW_hh   = (const float*)d_in[4];
  const float* gb_ih   = (const float*)d_in[5];
  const float* gb_hh   = (const float*)d_in[6];
  const float* cWq = (const float*)d_in[7];  const float* cbq = (const float*)d_in[8];
  const float* cWk = (const float*)d_in[9];  const float* cbk = (const float*)d_in[10];
  const float* cWv = (const float*)d_in[11]; const float* cbv = (const float*)d_in[12];
  const float* cWe = (const float*)d_in[13];
  const float* cWs = (const float*)d_in[14]; const float* cbs = (const float*)d_in[15];
  const float* oWq = (const float*)d_in[16]; const float* obq = (const float*)d_in[17];
  const float* oWk = (const float*)d_in[18]; const float* obk = (const float*)d_in[19];
  const float* oWv = (const float*)d_in[20]; const float* obv = (const float*)d_in[21];
  const float* oWe = (const float*)d_in[22];
  const float* oWs = (const float*)d_in[23]; const float* obs = (const float*)d_in[24];
  const float* aW  = (const float*)d_in[25]; const float* ab  = (const float*)d_in[26];
  float* out = (float*)d_out;

  // workspace bump allocator (256B aligned)
  char* ws = (char*)d_ws;
  size_t off = 0;
  auto alloc = [&](size_t bytes) -> void* {
    void* p = (void*)(ws + off);
    off += (bytes + 255) & ~(size_t)255;
    return p;
  };
  float*    h      = (float*)   alloc((size_t)N * H * 4);     // GRU hidden; reused as h_attn
  bf16_t*   hbf    = (bf16_t*)  alloc((size_t)N * H * 2);
  float*    gi     = (float*)   alloc((size_t)N * G * 4);     // reused as q|k|v
  float*    gh     = (float*)   alloc((size_t)N * G * 4);     // reused as agg|hcur
  float*    hstack = (float*)   alloc((size_t)T * N * H * 4);
  float*    alpha  = (float*)   alloc((size_t)E * 4);
  unsigned* enc    = (unsigned*)alloc((size_t)N * 4);
  float*    amax   = (float*)   alloc((size_t)N * 4);
  float*    denom  = (float*)   alloc((size_t)N * 4);
  bf16_t*   xbf    = (bf16_t*)  alloc((size_t)N * FIN * 2);
  bf16_t*   Wih    = (bf16_t*)  alloc((size_t)G * FIN * 2);
  bf16_t*   Whh    = (bf16_t*)  alloc((size_t)G * H * 2);
  bf16_t*   Wqb    = (bf16_t*)  alloc((size_t)NL * H * H * 2);
  bf16_t*   Wkb    = (bf16_t*)  alloc((size_t)NL * H * H * 2);
  bf16_t*   Wvb    = (bf16_t*)  alloc((size_t)NL * H * H * 2);
  bf16_t*   Wsb    = (bf16_t*)  alloc((size_t)NL * H * H * 2);
  (void)ws_size; (void)n_in; (void)out_size;

  auto cdiv = [](long long a, long long b) -> unsigned { return (unsigned)((a + b - 1) / b); };
  auto cvt = [&](const float* s, bf16_t* d, long long n) {
    to_bf16_kernel<<<cdiv(n, 256), 256, 0, stream>>>(s, d, n);
  };
  auto gemm = [&](const bf16_t* A, const bf16_t* W, const float* bias, float* C,
                  int M, int K, int Nout) {
    long long tiles = (long long)(M / 16) * (Nout / 64);   // 16x64 per wave
    wmma_gemm_bias<<<cdiv(tiles, 8), 256, 0, stream>>>(A, W, bias, C, M, K, Nout);
  };

  // weight conversions (every call: no cached state)
  cvt(gW_ih, Wih, (long long)G * FIN);
  cvt(gW_hh, Whh, (long long)G * H);
  cvt(cWq, Wqb, (long long)NL * H * H);
  cvt(cWk, Wkb, (long long)NL * H * H);
  cvt(cWv, Wvb, (long long)NL * H * H);
  cvt(cWs, Wsb, (long long)NL * H * H);

  // ---- GRU over T steps ----
  fill_kernel<<<cdiv((long long)N * H, 256), 256, 0, stream>>>(h, 0.0f, (long long)N * H);
  for (int t = 0; t < T; ++t) {
    cvt(x_seq + (size_t)t * N * FIN, xbf, (long long)N * FIN);
    cvt(h, hbf, (long long)N * H);
    gemm(xbf, Wih, gb_ih, gi, N, FIN, G);
    gemm(hbf, Whh, gb_hh, gh, N, H, G);
    gru_gate_kernel<<<cdiv((long long)N * H, 256), 256, 0, stream>>>(
        gi, gh, h, h, hstack + (size_t)t * N * H, N, H);
  }

  // ---- per-timestep 2-layer TransformerConv ----
  float* q = gi;  float* k = gi + (size_t)N * H;  float* v = gi + (size_t)2 * N * H;
  float* agg = gh; float* hcur = gh + (size_t)N * H;
  for (int t = 0; t < T; ++t) {
    const int* src = eis + (size_t)t * 2 * E;
    const int* dst = src + E;
    const float* ea = eas + (size_t)t * E;
    const float* hin = hstack + (size_t)t * N * H;
    for (int l = 0; l < NL; ++l) {
      cvt(hin, hbf, (long long)N * H);
      gemm(hbf, Wqb + (size_t)l * H * H, cbq + l * H, q,   N, H, H);
      gemm(hbf, Wkb + (size_t)l * H * H, cbk + l * H, k,   N, H, H);
      gemm(hbf, Wvb + (size_t)l * H * H, cbv + l * H, v,   N, H, H);
      gemm(hbf, Wsb + (size_t)l * H * H, cbs + l * H, agg, N, H, H);  // skip init
      init_nodes_kernel<<<cdiv(N, 256), 256, 0, stream>>>(enc, denom, N);
      edge_alpha_kernel<<<cdiv(E, 8), 256, 0, stream>>>(
          q, k, cWe + l * H, ea, src, dst, alpha, enc, E, H, scale);
      fix_amax_kernel<<<cdiv(N, 256), 256, 0, stream>>>(enc, amax, N);
      edge_exp_kernel<<<cdiv(E, 256), 256, 0, stream>>>(alpha, amax, denom, dst, E);
      edge_msg_kernel<<<cdiv(E, 8), 256, 0, stream>>>(
          alpha, denom, v, cWe + l * H, ea, src, dst, agg, E, H);
      float* hout = (l == NL - 1) ? (hstack + (size_t)t * N * H) : hcur;
      leaky_kernel<<<cdiv((long long)N * H, 256), 256, 0, stream>>>(agg, hout, (long long)N * H);
      hin = hout;
    }
  }

  // ---- temporal attention + top-3 sparse weighting ----
  float* hattn = h;  // GRU hidden no longer needed
  attn_kernel<<<cdiv(N, 8), 256, 0, stream>>>(hstack, aW, ab, hattn, N, H, T);

  // ---- output TransformerConv (128 -> 1) on last timestep edges ----
  const int* srcL = eis + (size_t)(T - 1) * 2 * E;
  const int* dstL = srcL + E;
  const float* eaL = eas + (size_t)(T - 1) * E;
  float* qs = gi; float* ks = gi + N; float* vs = gi + 2 * N;
  out_node_kernel<<<cdiv(N, 8), 256, 0, stream>>>(
      hattn, oWq, oWk, oWv, oWs, obq, obk, obv, obs, qs, ks, vs, out, N, H);
  init_nodes_kernel<<<cdiv(N, 256), 256, 0, stream>>>(enc, denom, N);
  out_edge_alpha_kernel<<<cdiv(E, 256), 256, 0, stream>>>(qs, ks, eaL, oWe, srcL, dstL, alpha, enc, E);
  fix_amax_kernel<<<cdiv(N, 256), 256, 0, stream>>>(enc, amax, N);
  edge_exp_kernel<<<cdiv(E, 256), 256, 0, stream>>>(alpha, amax, denom, dstL, E);
  out_edge_msg_kernel<<<cdiv(E, 256), 256, 0, stream>>>(alpha, denom, vs, eaL, oWe, srcL, dstL, out, E);
}
